// CRD_2576980377945
// MI455X (gfx1250) — matrix operand order repaired and checked by reading, not compiled, not run on hardware
//
#include <hip/hip_runtime.h>
#include <math.h>

// ---------------------------------------------------------------------------
// Contrastive (memory-bank / InfoNCE) loss + EMA bank update for MI455X.
//
// Bound by the random-row gather (≈1.07 GB @ 23.3 TB/s ≈ 46 µs) and the bank
// copy (≈1.23 GB ≈ 53 µs). Dot products run on the WMMA pipe
// (v_wmma_f32_16x16x32_f16, f32 accumulate). The gather uses CDNA5 async
// global->LDS loads (ASYNCcnt, saddr form with scalar row bases via
// v_readlane) software-pipelined one tile ahead so memory latency hides
// behind WMMA + f16 conversion work.
// ---------------------------------------------------------------------------

#define N_DATA   1200000
#define FEAT_DIM 128
#define BATCH    512
#define N_NEG    4096
#define TEMPV    0.07f
#define EPSN     1e-12f

typedef __attribute__((ext_vector_type(16))) _Float16 v16h;
typedef __attribute__((ext_vector_type(8)))  float    v8f;

// ---------------- Kernel 1: normalize s,t and positive logits ----------------
__global__ void crd_norm_pos(const float* __restrict__ stu,
                             const float* __restrict__ tea,
                             float* __restrict__ s_n,
                             float* __restrict__ t_n,
                             float* __restrict__ pos) {
  const int wave = threadIdx.x >> 5, lane = threadIdx.x & 31;
  const int row  = blockIdx.x * 8 + wave;
  if (row >= BATCH) return;

  const float4* sp = (const float4*)(stu + (size_t)row * FEAT_DIM);
  const float4* tp = (const float4*)(tea + (size_t)row * FEAT_DIM);
  float4 sv = sp[lane];
  float4 tv = tp[lane];

  float ss = sv.x*sv.x + sv.y*sv.y + sv.z*sv.z + sv.w*sv.w;
  float tt = tv.x*tv.x + tv.y*tv.y + tv.z*tv.z + tv.w*tv.w;
  #pragma unroll
  for (int off = 16; off; off >>= 1) {
    ss += __shfl_xor(ss, off);
    tt += __shfl_xor(tt, off);
  }
  const float sinv = 1.0f / fmaxf(sqrtf(ss), EPSN);
  const float tinv = 1.0f / fmaxf(sqrtf(tt), EPSN);

  float4 sn, tn;
  sn.x = sv.x*sinv; sn.y = sv.y*sinv; sn.z = sv.z*sinv; sn.w = sv.w*sinv;
  tn.x = tv.x*tinv; tn.y = tv.y*tinv; tn.z = tv.z*tinv; tn.w = tv.w*tinv;

  float p = sn.x*tn.x + sn.y*tn.y + sn.z*tn.z + sn.w*tn.w;
  #pragma unroll
  for (int off = 16; off; off >>= 1) p += __shfl_xor(p, off);

  ((float4*)(s_n + (size_t)row * FEAT_DIM))[lane] = sn;
  ((float4*)(t_n + (size_t)row * FEAT_DIM))[lane] = tn;
  if (lane == 0) pos[row] = p / TEMPV;
}

// ---------------- Kernel 2: negative logits via WMMA + async gather ----------
// Block = 128 threads (4 waves), one batch row b per block, 256 negatives per
// block (each wave: 4 tiles of 16 negatives). Pipeline per tile:
//   s_wait_asynccnt 0             (gather of tile t landed in LDS)
//   16x ds_load_b128 -> st[16]    (batched; one wait, no serialized stalls)
//   s_wait_dscnt 0                (LDS buffer free for reuse)
//   issue async gather of t+1     (saddr form: scalar row base + lane*16)
//   f16 convert + self-dot + 4x v_wmma_f32_16x16x32_f16 on tile t
__global__ void __launch_bounds__(128)
crd_neg_logits(const float* __restrict__ mem,
               const float* __restrict__ s_n,
               const int*   __restrict__ indices,
               const int*   __restrict__ rneg,
               float* __restrict__ neg) {
  __shared__ float s_vec[FEAT_DIM];
  __shared__ float rows[4][16 * 132];   // per-wave staging, padded stride

  const int b    = blockIdx.y;
  const int k0   = blockIdx.x * 256;
  const int w    = threadIdx.x >> 5;
  const int lane = threadIdx.x & 31;
  const int hi   = lane >> 4;           // 0: lanes 0-15, 1: lanes 16-31
  const int m    = lane & 15;           // negative row within tile

  if (threadIdx.x < FEAT_DIM) s_vec[threadIdx.x] = s_n[(size_t)b * FEAT_DIM + threadIdx.x];
  __syncthreads();

  // B operand (32x16 f16 per chunk, all 16 columns = s[b]).
  // Layout: VGPR v holds K = {2v, 2v+1} (v<4) / {2v+8, 2v+9} (v>=4) for lanes 0-15, K+8 for lanes 16-31.
  v16h bop[4];
  #pragma unroll
  for (int c = 0; c < 4; ++c) {
    #pragma unroll
    for (int v = 0; v < 8; ++v) {
      const int K = c * 32 + hi * 8 + 2 * v + (v >= 4 ? 8 : 0);
      bop[c][2 * v]     = (_Float16)s_vec[K];
      bop[c][2 * v + 1] = (_Float16)s_vec[K + 1];
    }
  }

  // Shifted negative indices for this wave's 4 tiles (held by lanes 0-15).
  const int ib = indices[b];
  int nidx[4] = {0, 0, 0, 0};
  if (lane < 16) {
    #pragma unroll
    for (int t = 0; t < 4; ++t) {
      const int kk = k0 + w * 64 + t * 16 + lane;
      const int rr = rneg[(size_t)b * N_NEG + kk];
      nidx[t] = rr + (rr >= ib ? 1 : 0);
    }
  }

  // Per-lane constants for the async gather (GVS addressing):
  //   mem_addr = SGPR(row base) + voff,  lds_addr = ldslane + j*528
  const unsigned voff    = (unsigned)lane * 16u;
  const unsigned ldslane = (unsigned)(uintptr_t)&rows[w][0] + voff;

  // Prologue: async gather of tile 0 (each j: 32 lanes cover one 512B row).
  #pragma unroll
  for (int j = 0; j < 16; ++j) {
    const int ridx = __builtin_amdgcn_readlane(nidx[0], j);       // SGPR
    const float* base = mem + (size_t)ridx * FEAT_DIM;            // scalar math
    const unsigned ldst = ldslane + (unsigned)(j * 528);
    asm volatile("global_load_async_to_lds_b128 %0, %1, %2"
                 :: "v"(ldst), "v"(voff), "s"(base) : "memory");
  }

  #pragma unroll
  for (int t = 0; t < 4; ++t) {
    // gather of tile t complete in LDS
    asm volatile("s_wait_asynccnt 0x0" ::: "memory");

    // batched LDS reads for this lane's negative row (row m, K halves by hi)
    const float* rp = &rows[w][m * 132];
    float4 st[16];
    #pragma unroll
    for (int c = 0; c < 4; ++c) {
      const int base = c * 32 + hi * 8;
      st[4 * c + 0] = *(const float4*)&rp[base];
      st[4 * c + 1] = *(const float4*)&rp[base + 4];
      st[4 * c + 2] = *(const float4*)&rp[base + 16];
      st[4 * c + 3] = *(const float4*)&rp[base + 20];
    }
    // LDS buffer values are in registers; safe to overwrite the buffer
    asm volatile("s_wait_dscnt 0x0" ::: "memory");

    // issue async gather for tile t+1 (overlaps with WMMA below)
    if (t < 3) {
      #pragma unroll
      for (int j = 0; j < 16; ++j) {
        const int ridx = __builtin_amdgcn_readlane(nidx[t + 1], j);
        const float* base = mem + (size_t)ridx * FEAT_DIM;
        const unsigned ldst = ldslane + (unsigned)(j * 528);
        asm volatile("global_load_async_to_lds_b128 %0, %1, %2"
                     :: "v"(ldst), "v"(voff), "s"(base) : "memory");
      }
    }

    // A build (documented 16-bit A layout) + WMMA accumulate + self-dot
    v8f acc = {};
    float selfacc = 0.0f;
    #pragma unroll
    for (int c = 0; c < 4; ++c) {
      const float4 q0 = st[4 * c + 0];
      const float4 q1 = st[4 * c + 1];
      const float4 q2 = st[4 * c + 2];
      const float4 q3 = st[4 * c + 3];
      v16h a;
      a[0]  = (_Float16)q0.x; a[1]  = (_Float16)q0.y;
      a[2]  = (_Float16)q0.z; a[3]  = (_Float16)q0.w;
      a[4]  = (_Float16)q1.x; a[5]  = (_Float16)q1.y;
      a[6]  = (_Float16)q1.z; a[7]  = (_Float16)q1.w;
      a[8]  = (_Float16)q2.x; a[9]  = (_Float16)q2.y;
      a[10] = (_Float16)q2.z; a[11] = (_Float16)q2.w;
      a[12] = (_Float16)q3.x; a[13] = (_Float16)q3.y;
      a[14] = (_Float16)q3.z; a[15] = (_Float16)q3.w;
      selfacc += q0.x*q0.x + q0.y*q0.y + q0.z*q0.z + q0.w*q0.w
               + q1.x*q1.x + q1.y*q1.y + q1.z*q1.z + q1.w*q1.w
               + q2.x*q2.x + q2.y*q2.y + q2.z*q2.z + q2.w*q2.w
               + q3.x*q3.x + q3.y*q3.y + q3.z*q3.z + q3.w*q3.w;
      acc = __builtin_amdgcn_wmma_f32_16x16x32_f16(
          /*neg_a=*/false, a, /*neg_b=*/false, bop[c],
          /*c_mod=*/(short)0, acc, /*reuse_a=*/false, /*reuse_b=*/false);
    }
    // lanes (m) and (m+16) hold complementary K halves of row m's self-dot
    const float full = selfacc + __shfl_xor(selfacc, 16);

    // D layout: vgpr i at lanes 0-15 -> neg i (all cols equal),
    //           vgpr i at lanes 16-31 -> neg i+8.
    const int kk0 = k0 + w * 64 + t * 16;
    #pragma unroll
    for (int i = 0; i < 8; ++i) {
      const float sd0 = __shfl(full, i);       // self-dot of neg i
      const float sd1 = __shfl(full, i + 8);   // self-dot of neg i+8
      if (lane == 0)
        neg[(size_t)b * N_NEG + kk0 + i] =
            acc[i] / fmaxf(sqrtf(sd0), EPSN) / TEMPV;
      if (lane == 16)
        neg[(size_t)b * N_NEG + kk0 + 8 + i] =
            acc[i] / fmaxf(sqrtf(sd1), EPSN) / TEMPV;
    }
  }
}

// ---------------- Kernel 3: per-row logsumexp - pos --------------------------
__global__ void crd_lse(const float* __restrict__ neg,
                        const float* __restrict__ pos,
                        float* __restrict__ rowloss) {
  __shared__ float red[256];
  const int b = blockIdx.x, tid = threadIdx.x;
  const float* nb = neg + (size_t)b * N_NEG;
  const float p = pos[b];

  float lmax = -3.4e38f;
  for (int k = tid; k < N_NEG; k += 256) lmax = fmaxf(lmax, nb[k]);
  if (tid == 0) lmax = fmaxf(lmax, p);
  red[tid] = lmax; __syncthreads();
  for (int s = 128; s; s >>= 1) {
    if (tid < s) red[tid] = fmaxf(red[tid], red[tid + s]);
    __syncthreads();
  }
  const float M = red[0]; __syncthreads();

  float ssum = 0.0f;
  for (int k = tid; k < N_NEG; k += 256) ssum += expf(nb[k] - M);
  if (tid == 0) ssum += expf(p - M);
  red[tid] = ssum; __syncthreads();
  for (int s = 128; s; s >>= 1) {
    if (tid < s) red[tid] += red[tid + s];
    __syncthreads();
  }
  if (tid == 0) rowloss[b] = M + logf(red[0]) - p;
}

// ---------------- Kernel 4: mean over batch -> loss --------------------------
__global__ void crd_mean(const float* __restrict__ rowloss,
                         float* __restrict__ out) {
  __shared__ float red[512];
  const int tid = threadIdx.x;
  red[tid] = rowloss[tid]; __syncthreads();
  for (int s = 256; s; s >>= 1) {
    if (tid < s) red[tid] += red[tid + s];
    __syncthreads();
  }
  if (tid == 0) out[0] = red[0] * (1.0f / BATCH);
}

// ---------------- Kernel 5: EMA scatter update of copied bank ----------------
__global__ void crd_scatter(const float* __restrict__ mem,
                            const float* __restrict__ t_n,
                            const int*   __restrict__ indices,
                            float* __restrict__ outbank) {
  __shared__ float red[128];
  const int b = blockIdx.x, tid = threadIdx.x;
  const int row = indices[b];
  const float oldv = mem[(size_t)row * FEAT_DIM + tid];
  const float u = 0.5f * oldv + 0.5f * t_n[(size_t)b * FEAT_DIM + tid];
  red[tid] = u * u; __syncthreads();
  for (int s = 64; s; s >>= 1) {
    if (tid < s) red[tid] += red[tid + s];
    __syncthreads();
  }
  const float inv = 1.0f / fmaxf(sqrtf(red[0]), EPSN);
  outbank[(size_t)row * FEAT_DIM + tid] = u * inv;
}

// ---------------------------------------------------------------------------
extern "C" void kernel_launch(void* const* d_in, const int* in_sizes, int n_in,
                              void* d_out, int out_size, void* d_ws, size_t ws_size,
                              hipStream_t stream) {
  (void)in_sizes; (void)n_in; (void)out_size; (void)ws_size;

  const float* stu     = (const float*)d_in[0];  // [512,128]
  const float* tea     = (const float*)d_in[1];  // [512,128]
  const float* mem     = (const float*)d_in[2];  // [1.2M,128]
  const int*   indices = (const int*)  d_in[3];  // [512]
  const int*   rneg    = (const int*)  d_in[4];  // [512,4096]

  float* out = (float*)d_out;                    // [0]=loss, [1:]=new bank
  float* ws  = (float*)d_ws;
  float* s_n     = ws;                               // 512*128
  float* t_n     = s_n + (size_t)BATCH * FEAT_DIM;   // 512*128
  float* pos     = t_n + (size_t)BATCH * FEAT_DIM;   // 512
  float* neg     = pos + BATCH;                      // 512*4096
  float* rowloss = neg + (size_t)BATCH * N_NEG;      // 512

  // Bank copy first (longest transfer): 614 MB D2D via the copy path.
  hipMemcpyAsync(out + 1, mem, (size_t)N_DATA * FEAT_DIM * sizeof(float),
                 hipMemcpyDeviceToDevice, stream);

  crd_norm_pos<<<BATCH / 8, 256, 0, stream>>>(stu, tea, s_n, t_n, pos);

  dim3 g2(N_NEG / 256, BATCH);
  crd_neg_logits<<<g2, 128, 0, stream>>>(mem, s_n, indices, rneg, neg);

  crd_lse<<<BATCH, 256, 0, stream>>>(neg, pos, rowloss);
  crd_mean<<<1, 512, 0, stream>>>(rowloss, out);

  crd_scatter<<<BATCH, 128, 0, stream>>>(mem, t_n, indices, out + 1);
}
